// Net_8555574854363
// MI455X (gfx1250) — compile-verified, matrix operand order checked by reading, not run on hardware
//
#include <hip/hip_runtime.h>
#include <math.h>

typedef __attribute__((ext_vector_type(16))) __bf16 v16bf;
typedef __attribute__((ext_vector_type(8)))  __bf16 v8bf;
typedef __attribute__((ext_vector_type(8)))  float  v8f;
typedef __attribute__((ext_vector_type(8)))  unsigned v8u;
typedef int v4i_ __attribute__((vector_size(16)));   // matches async-LDS builtin pointee

// ---------- helpers ----------

// f32 -> bf16, round-half-up (1 add + high-half extract).
__device__ __forceinline__ __bf16 f2bf(float f) {
  unsigned u = __builtin_bit_cast(unsigned, f) + 0x8000u;
  return __builtin_bit_cast(__bf16, (unsigned short)(u >> 16));
}

// Pack two f32 into two bf16 in one dword: 2x v_add + 1x v_perm_b32.
// result lo16 = bf16(x), hi16 = bf16(y).
__device__ __forceinline__ unsigned pack2bf(float x, float y) {
  unsigned ux = __builtin_bit_cast(unsigned, x) + 0x8000u;
  unsigned uy = __builtin_bit_cast(unsigned, y) + 0x8000u;
  return __builtin_amdgcn_perm(uy, ux, 0x07060302u);  // [uy.b3 uy.b2 ux.b3 ux.b2]
}

__device__ __forceinline__ float sigmoidf_(float x) {
  return 1.0f / (1.0f + __expf(-x));
}

// ---------- async memory -> LDS staging (gfx1250 GLOBAL_LOAD_ASYNC_TO_LDS) ----------
#if __has_builtin(__builtin_amdgcn_global_load_async_to_lds_b128)
#define HAVE_ASYNC_LDS 1
__device__ __forceinline__ void stage_cp16(const void* g, void* l) {
  __builtin_amdgcn_global_load_async_to_lds_b128(
      (__attribute__((address_space(1))) v4i_*)(unsigned long long)g,
      (__attribute__((address_space(3))) v4i_*)(unsigned)(unsigned long long)l,
      0, 0);
}
#else
#define HAVE_ASYNC_LDS 0
__device__ __forceinline__ void stage_cp16(const void* g, void* l) {
  *(uint4*)l = *(const uint4*)g;
}
#endif

__device__ __forceinline__ void stage_wait() {
#if HAVE_ASYNC_LDS
#if __has_builtin(__builtin_amdgcn_s_wait_asynccnt)
  __builtin_amdgcn_s_wait_asynccnt(0);
#else
  asm volatile("s_wait_asynccnt 0" ::: "memory");
#endif
#endif
}

// A-fragment (16x32 bf16) from fp32 row-major, inline perm-pack conversion.
// elem i -> K = (i<8 ? i : i+8) + (lane>=16 ? 8 : 0); M = lane%16.
__device__ __forceinline__ v16bf load_A_frag_f32(const float* __restrict__ base, int ld,
                                                 int row0, int kb, int lane) {
  const int m   = lane & 15;
  const int khi = (lane >> 4) << 3;          // 0 or 8
  const float* p = base + (size_t)(row0 + m) * ld + kb + khi;
  float4 f0 = *(const float4*)(p);           // K = khi+0..3
  float4 f1 = *(const float4*)(p + 4);       // K = khi+4..7
  float4 f2 = *(const float4*)(p + 16);      // K = khi+16..19
  float4 f3 = *(const float4*)(p + 20);      // K = khi+20..23
  v8u pk;
  pk[0] = pack2bf(f0.x, f0.y);  pk[1] = pack2bf(f0.z, f0.w);
  pk[2] = pack2bf(f1.x, f1.y);  pk[3] = pack2bf(f1.z, f1.w);
  pk[4] = pack2bf(f2.x, f2.y);  pk[5] = pack2bf(f2.z, f2.w);
  pk[6] = pack2bf(f3.x, f3.y);  pk[7] = pack2bf(f3.z, f3.w);
  return __builtin_bit_cast(v16bf, pk);
}

// A-fragment from bf16 row-major mirror: two 16B loads, zero conversion VALU.
__device__ __forceinline__ v16bf load_A_frag_bf16(const __bf16* __restrict__ base, int ld,
                                                  int row0, int kb, int lane) {
  const int m   = lane & 15;
  const int khi = (lane >> 4) << 3;
  const __bf16* p = base + (size_t)(row0 + m) * ld + kb + khi;
  v8bf lo = *(const v8bf*)p;          // K = khi .. khi+7
  v8bf hi = *(const v8bf*)(p + 16);   // K = khi+16 .. khi+23
  return __builtin_shufflevector(lo, hi, 0, 1, 2, 3, 4, 5, 6, 7,
                                 8, 9, 10, 11, 12, 13, 14, 15);
}

// B fragment read (LDS-staged): lane's 16 bf16 contiguous -> ds_load_b128 x2.
__device__ __forceinline__ v16bf load_B_frag(const __bf16* frags, int tile, int lane) {
  return *(const v16bf*)(frags + (size_t)tile * 512 + lane * 16);
}

__device__ __forceinline__ v8f wmma_bf16(v16bf a, v16bf b, v8f c) {
  return __builtin_amdgcn_wmma_f32_16x16x32_bf16(false, a, false, b, (short)0, c,
                                                 false, false);
}

// ---------- weight pre-pack into B-fragment layout ----------
// B_logical[k][n]: trans==0 -> src[k*ld+n]; trans==1 -> src[n*ld+k].
__global__ void pack_b_frag(const float* __restrict__ src, __bf16* __restrict__ dst,
                            int K, int Ncol, int ld, int trans) {
  int total = (K >> 5) * (Ncol >> 4) * 512;
  int nct = Ncol >> 4;
  for (int o = blockIdx.x * blockDim.x + threadIdx.x; o < total;
       o += gridDim.x * blockDim.x) {
    int tile = o >> 9;
    int within = o & 511;
    int lane = within >> 4;
    int i = within & 15;
    int t = tile / nct, c = tile % nct;
    int k = 32 * t + ((i < 8) ? i : i + 8) + ((lane >= 16) ? 8 : 0);
    int n = 16 * c + (lane & 15);
    float v = trans ? src[(size_t)n * ld + k] : src[(size_t)k * ld + n];
    dst[o] = f2bf(v);
  }
}

__global__ void zero_f32v4(float4* __restrict__ p, long long n4) {
  long long i = (long long)blockIdx.x * blockDim.x + threadIdx.x;
  if (i < n4) p[i] = make_float4(0.f, 0.f, 0.f, 0.f);
}

__global__ void cvt_i64_i32(const long long* __restrict__ src, int* __restrict__ dst,
                            long long n) {
  long long i = (long long)blockIdx.x * blockDim.x + threadIdx.x;
  if (i < n) dst[i] = (int)src[i];
}

// ---------- reduce: h = x @ W_red^T + b_red  ([N,256] x [256,64]) ----------
__global__ void reduce_gemm(const float* __restrict__ x, const __bf16* __restrict__ WredF,
                            const float* __restrict__ b_red, float* __restrict__ h,
                            __bf16* __restrict__ hb, int ntiles) {
  __shared__ __bf16 smB[32 * 512];                       // 32 KB weight fragments
  for (int i = threadIdx.x; i < 32 * 512 / 8; i += blockDim.x)
    stage_cp16((const uint4*)WredF + i, (uint4*)smB + i);
  stage_wait();
  __syncthreads();

  int wave = (blockIdx.x * blockDim.x + threadIdx.x) >> 5;
  int lane = threadIdx.x & 31;
  if (wave >= ntiles) return;
  int row0 = wave * 16;

  const float* prow = x + (size_t)(row0 + (lane & 15)) * 256;
  v8f acc[4] = {{}, {}, {}, {}};
#pragma unroll
  for (int t = 0; t < 8; ++t) {
    if (t < 6) __builtin_prefetch(prow + 32 * (t + 2), 0, 1);   // global_prefetch_b8
    v16bf a = load_A_frag_f32(x, 256, row0, 32 * t, lane);
#pragma unroll
    for (int c = 0; c < 4; ++c)
      acc[c] = wmma_bf16(a, load_B_frag(smB, t * 4 + c, lane), acc[c]);
  }
  int colb = lane & 15;
  int rbase = row0 + ((lane >> 4) << 3);
#pragma unroll
  for (int r = 0; r < 8; ++r) {
    size_t ro = (size_t)(rbase + r) * 64;
#pragma unroll
    for (int c = 0; c < 4; ++c) {
      float v = acc[c][r] + b_red[16 * c + colb];
      h[ro + 16 * c + colb] = v;
      hb[ro + 16 * c + colb] = f2bf(v);
    }
  }
}

// ---------- per-step transform: m = hb @ W_l  ([N,64] x [64,64]) ----------
__global__ void msg_gemm(const __bf16* __restrict__ hb, const __bf16* __restrict__ WlF,
                         float* __restrict__ m, int ntiles) {
  __shared__ __bf16 smB[8 * 512];                        // 8 KB
  for (int i = threadIdx.x; i < 8 * 512 / 8; i += blockDim.x)
    stage_cp16((const uint4*)WlF + i, (uint4*)smB + i);
  stage_wait();
  __syncthreads();

  int wave = (blockIdx.x * blockDim.x + threadIdx.x) >> 5;
  int lane = threadIdx.x & 31;
  if (wave >= ntiles) return;
  int row0 = wave * 16;

  v8f acc[4] = {{}, {}, {}, {}};
#pragma unroll
  for (int t = 0; t < 2; ++t) {
    v16bf a = load_A_frag_bf16(hb, 64, row0, 32 * t, lane);
#pragma unroll
    for (int c = 0; c < 4; ++c)
      acc[c] = wmma_bf16(a, load_B_frag(smB, t * 4 + c, lane), acc[c]);
  }
  int colb = lane & 15;
  int rbase = row0 + ((lane >> 4) << 3);
#pragma unroll
  for (int r = 0; r < 8; ++r) {
    size_t ro = (size_t)(rbase + r) * 64;
#pragma unroll
    for (int c = 0; c < 4; ++c)
      m[ro + 16 * c + colb] = acc[c][r];
  }
}

// ---------- edge scatter: agg[dst] += m[src] (fp32 atomics, L2-resident) ----------
__global__ void scatter_add(const float* __restrict__ m, const int* __restrict__ src,
                            const int* __restrict__ dst, float* __restrict__ agg,
                            int E) {
  long long i = (long long)blockIdx.x * blockDim.x + threadIdx.x;
  long long total = (long long)E * 16;                   // float4 granules
  if (i >= total) return;
  int e = (int)(i >> 4);
  int q = (int)(i & 15);
  int s = src[e];
  int d = dst[e];
  float4 v = ((const float4*)(m + (size_t)s * 64))[q];
  float* a = agg + (size_t)d * 64 + q * 4;
  atomicAdd(a + 0, v.x);
  atomicAdd(a + 1, v.y);
  atomicAdd(a + 2, v.z);
  atomicAdd(a + 3, v.w);
}

// ---------- fused GRU ----------
// r/z gates chain agg@W_ih and h@W_hh WMMAs into one accumulator.
__global__ void gru_gemm(const float* __restrict__ agg, const float* __restrict__ h_in,
                         const __bf16* __restrict__ hb_in,
                         const __bf16* __restrict__ WihF, const __bf16* __restrict__ WhhF,
                         const float* __restrict__ b_ih, const float* __restrict__ b_hh,
                         float* __restrict__ h_out, __bf16* __restrict__ hb_out,
                         int ntiles) {
  __shared__ __bf16 smB[48 * 512];                       // 24 KB Wih + 24 KB Whh
  {
    const int n16 = 24 * 512 / 8;
    for (int i = threadIdx.x; i < n16; i += blockDim.x)
      stage_cp16((const uint4*)WihF + i, (uint4*)smB + i);
    for (int i = threadIdx.x; i < n16; i += blockDim.x)
      stage_cp16((const uint4*)WhhF + i, (uint4*)smB + n16 + i);
  }
  stage_wait();
  __syncthreads();

  int wave = (blockIdx.x * blockDim.x + threadIdx.x) >> 5;
  int lane = threadIdx.x & 31;
  if (wave >= ntiles) return;
  int row0 = wave * 16;

  v16bf Aagg[2], Ah[2];
#pragma unroll
  for (int t = 0; t < 2; ++t) {
    Aagg[t] = load_A_frag_f32(agg, 64, row0, 32 * t, lane);    // read-once: inline cvt
    Ah[t]   = load_A_frag_bf16(hb_in, 64, row0, 32 * t, lane); // mirror: pure loads
  }

  int colb = lane & 15;
  int rbase = row0 + ((lane >> 4) << 3);

#pragma unroll
  for (int c = 0; c < 4; ++c) {
    v8f ar = {}, az = {}, ain = {}, ahn = {};
#pragma unroll
    for (int t = 0; t < 2; ++t) {
      ar  = wmma_bf16(Aagg[t], load_B_frag(smB, t * 12 + c,          lane), ar);
      ar  = wmma_bf16(Ah[t],   load_B_frag(smB, 24 + t * 12 + c,     lane), ar);
      az  = wmma_bf16(Aagg[t], load_B_frag(smB, t * 12 + c + 4,      lane), az);
      az  = wmma_bf16(Ah[t],   load_B_frag(smB, 24 + t * 12 + c + 4, lane), az);
      ain = wmma_bf16(Aagg[t], load_B_frag(smB, t * 12 + c + 8,      lane), ain);
      ahn = wmma_bf16(Ah[t],   load_B_frag(smB, 24 + t * 12 + c + 8, lane), ahn);
    }
    int col = 16 * c + colb;
    float brs = b_ih[col]      + b_hh[col];
    float bzs = b_ih[64 + col] + b_hh[64 + col];
    float bin = b_ih[128 + col];
    float bhn = b_hh[128 + col];
#pragma unroll
    for (int r = 0; r < 8; ++r) {
      size_t ro = (size_t)(rbase + r) * 64;
      float hold = h_in[ro + col];
      float rg = sigmoidf_(ar[r] + brs);
      float zg = sigmoidf_(az[r] + bzs);
      float ng = tanhf(ain[r] + bin + rg * (ahn[r] + bhn));
      float hn = (1.0f - zg) * ng + zg * hold;
      h_out[ro + col] = hn;
      hb_out[ro + col] = f2bf(hn);
    }
  }
}

// ---------- head: out[s] = sigmoid( dot(sigmoid(h[idx[s]]), W_lin) + b_lin ) ----------
__global__ void head_kernel(const float* __restrict__ h, const long long* __restrict__ idx,
                            const float* __restrict__ Wlin, const float* __restrict__ blin,
                            float* __restrict__ out, int S) {
  int s = blockIdx.x * blockDim.x + threadIdx.x;
  if (s >= S) return;
  int node = (int)idx[s];
  const float4* hp = (const float4*)(h + (size_t)node * 64);
  const float4* wp = (const float4*)Wlin;
  float acc = 0.0f;
#pragma unroll
  for (int q = 0; q < 16; ++q) {
    float4 v = hp[q];
    float4 w = wp[q];
    acc += sigmoidf_(v.x) * w.x + sigmoidf_(v.y) * w.y +
           sigmoidf_(v.z) * w.z + sigmoidf_(v.w) * w.w;
  }
  out[s] = sigmoidf_(acc + blin[0]);
}

// ---------- host ----------
extern "C" void kernel_launch(void* const* d_in, const int* in_sizes, int n_in,
                              void* d_out, int out_size, void* d_ws, size_t ws_size,
                              hipStream_t stream) {
  const float*     x     = (const float*)d_in[0];
  const long long* eidx  = (const long long*)d_in[1];
  const long long* sel   = (const long long*)d_in[2];
  const float*     W_red = (const float*)d_in[3];
  const float*     b_red = (const float*)d_in[4];
  const float*     ggc_W = (const float*)d_in[5];
  const float*     W_ih  = (const float*)d_in[6];
  const float*     W_hh  = (const float*)d_in[7];
  const float*     b_ih  = (const float*)d_in[8];
  const float*     b_hh  = (const float*)d_in[9];
  const float*     W_lin = (const float*)d_in[10];
  const float*     b_lin = (const float*)d_in[11];
  float* out = (float*)d_out;

  const int A = 256, H = 64, L = 8;
  const int N = in_sizes[0] / A;
  const int E = in_sizes[1] / 2;
  const int S = in_sizes[2];

  char* ws = (char*)d_ws;
  size_t off = 0;
  auto carve = [&](size_t bytes) -> void* {
    void* p = ws + off;
    off = (off + bytes + 255) & ~(size_t)255;
    return p;
  };
  float*  hA    = (float*)carve((size_t)N * H * 4);
  float*  hB    = (float*)carve((size_t)N * H * 4);
  __bf16* hAb   = (__bf16*)carve((size_t)N * H * 2);
  __bf16* hBb   = (__bf16*)carve((size_t)N * H * 2);
  float*  msg   = (float*)carve((size_t)N * H * 4);
  float*  agg   = (float*)carve((size_t)N * H * 4);
  int*    e32   = (int*)carve((size_t)2 * E * 4);
  __bf16* WredF = (__bf16*)carve((size_t)8 * 4 * 512 * 2);
  __bf16* ggcF  = (__bf16*)carve((size_t)L * 2 * 4 * 512 * 2);
  __bf16* WihF  = (__bf16*)carve((size_t)2 * 12 * 512 * 2);
  __bf16* WhhF  = (__bf16*)carve((size_t)2 * 12 * 512 * 2);
  (void)ws_size; (void)n_in; (void)out_size;

  // One-time prep: weight fragments + int32 edge indices.
  pack_b_frag<<<64, 256, 0, stream>>>(W_red, WredF, A, H, A, 1);
  for (int l = 0; l < L; ++l)
    pack_b_frag<<<32, 256, 0, stream>>>(ggc_W + (size_t)l * H * H,
                                        ggcF + (size_t)l * 2 * 4 * 512, H, H, H, 0);
  pack_b_frag<<<48, 256, 0, stream>>>(W_ih, WihF, H, 3 * H, H, 1);
  pack_b_frag<<<48, 256, 0, stream>>>(W_hh, WhhF, H, 3 * H, H, 1);
  cvt_i64_i32<<<(int)((2LL * E + 255) / 256), 256, 0, stream>>>(eidx, e32, 2LL * E);

  int ntiles = N / 16;                       // N = 100000 -> 6250 exact
  int gblocks = (ntiles + 7) / 8;            // 8 wave32 waves per 256-thread block

  reduce_gemm<<<gblocks, 256, 0, stream>>>(x, WredF, b_red, hA, hAb, ntiles);

  long long nh4 = (long long)N * H / 4;
  int zblocks = (int)((nh4 + 255) / 256);
  long long tsc = (long long)E * 16;
  int sblocks = (int)((tsc + 255) / 256);

  float*  hcur = hA,  *hnxt = hB;
  __bf16* hbcur = hAb, *hbnxt = hBb;
  for (int l = 0; l < L; ++l) {
    msg_gemm<<<gblocks, 256, 0, stream>>>(hbcur, ggcF + (size_t)l * 2 * 4 * 512, msg, ntiles);
    zero_f32v4<<<zblocks, 256, 0, stream>>>((float4*)agg, nh4);
    scatter_add<<<sblocks, 256, 0, stream>>>(msg, e32, e32 + E, agg, E);
    gru_gemm<<<gblocks, 256, 0, stream>>>(agg, hcur, hbcur, WihF, WhhF, b_ih, b_hh,
                                          hnxt, hbnxt, ntiles);
    float* tf = hcur; hcur = hnxt; hnxt = tf;
    __bf16* tb = hbcur; hbcur = hbnxt; hbnxt = tb;
  }

  head_kernel<<<(S + 255) / 256, 256, 0, stream>>>(hcur, sel, W_lin, b_lin, out, S);
}
